// RNN_51994874085959
// MI455X (gfx1250) — matrix operand order repaired and checked by reading, not compiled, not run on hardware
//
#include <hip/hip_runtime.h>
#include <hip/hip_bf16.h>

typedef __attribute__((ext_vector_type(16))) __bf16 v16bf;
typedef __attribute__((ext_vector_type(8)))  float  v8f;

#define T_DIM 512
#define B_DIM 1024
#define H_DIM 64
#define O_DIM 12
#define G_DIM 256 /* 4*H */

static __device__ __forceinline__ float sigf(float x) {
  return 1.0f / (1.0f + __expf(-x));
}

// Pack 4x float4 (two contiguous 8-float runs at K and K+16) into the
// 16x32 bf16 A-operand lane layout (per-lane 16 bf16 elements).
static __device__ __forceinline__ v16bf packA(float4 a, float4 b, float4 c, float4 d) {
  v16bf r;
  r[0]  = (__bf16)a.x; r[1]  = (__bf16)a.y; r[2]  = (__bf16)a.z; r[3]  = (__bf16)a.w;
  r[4]  = (__bf16)b.x; r[5]  = (__bf16)b.y; r[6]  = (__bf16)b.z; r[7]  = (__bf16)b.w;
  r[8]  = (__bf16)c.x; r[9]  = (__bf16)c.y; r[10] = (__bf16)c.z; r[11] = (__bf16)c.w;
  r[12] = (__bf16)d.x; r[13] = (__bf16)d.y; r[14] = (__bf16)d.z; r[15] = (__bf16)d.w;
  return r;
}

static __device__ __forceinline__ v8f wmma_bf16(v16bf a, v16bf b, v8f c) {
  // (neg_a, A, neg_b, B, c_mod, C, reuse_a, reuse_b)
  return __builtin_amdgcn_wmma_f32_16x16x32_bf16(false, a, false, b, (short)0, c,
                                                 false, false);
}

// ---------------------------------------------------------------------------
// Pack a row-major (nrows x 64) fp32 weight matrix W into the 32x16 bf16
// B-operand layout for g = x @ W^T:  B[k][n] = W[n][k].
// dst layout: [ct][kc][lane][e] flattened; n = ct*16 + lane%16,
//             k = kc*32 + (lane/16)*8 + (e%8) + (e/8)*16.  Rows beyond nrows pad to zero.
// ---------------------------------------------------------------------------
__global__ void pack_b_kernel(const float* __restrict__ src, __bf16* __restrict__ dst,
                              int nrows, int ct_count) {
  int idx = blockIdx.x * 256 + threadIdx.x;
  int total = ct_count * 1024;
  if (idx >= total) return;
  int e  = idx & 15;
  int l  = (idx >> 4) & 31;
  int kc = (idx >> 9) & 1;
  int ct = idx >> 10;
  int n = ct * 16 + (l & 15);
  int k = kc * 32 + (l >> 4) * 8 + (e & 7) + (e >> 3) * 16;
  float v = (n < nrows) ? src[n * 64 + k] : 0.0f;
  dst[idx] = (__bf16)v;
}

__global__ void bias_sum_kernel(const float* __restrict__ bih,
                                const float* __restrict__ bhh,
                                float* __restrict__ bsum) {
  int i = blockIdx.x * 256 + threadIdx.x;
  if (i < 4 * G_DIM) bsum[i] = bih[i] + bhh[i];
}

// ---------------------------------------------------------------------------
// Fused dilated-LSTM scan.  Effective batch N = dil*B (DRNN fold:
// row r = j*B + b covers times t = tp*dil + j).  Each wave owns 16 rows for
// the whole time loop: c-state lives in VGPRs, Wih/Whh live in LDS (bf16,
// B-operand layout), h round-trips through a per-wave LDS tile.
// Per step per wave: 64 v_wmma_f32_16x16x32_bf16 (16 col-tiles x 2 K x 2 mats).
// Gate chains start from an inline-0 C operand (SRC2=0, register-free) and the
// bias is added in the gate epilogue — avoids bias-splat copies and spills
// under the __launch_bounds__(256,4) VGPR cap.
// ---------------------------------------------------------------------------
__global__ __launch_bounds__(256, 4)
void lstm_scan_kernel(const float* __restrict__ xin,   // (T,B,64)
                      const float* __restrict__ resid, // null or (T,B,64)
                      float* __restrict__ hout,        // (T,B,64)
                      const __bf16* __restrict__ WihP, // [16][2][32][16]
                      const __bf16* __restrict__ WhhP, // [16][2][32][16]
                      const float* __restrict__ bias,  // 256 (bih+bhh)
                      int dil) {
  __shared__ alignas(32) __bf16 sW[2][16][2][32][16]; // 64 KB
  __shared__ alignas(16) float  hbuf[8][16][64];      // 32 KB (per-wave tiles)

  const int tid = threadIdx.x;
  { // cooperative weight load: 2 x 32 KB
    uint4* dst = (uint4*)&sW[0][0][0][0][0];
    const uint4* s0 = (const uint4*)WihP;
    const uint4* s1 = (const uint4*)WhhP;
#pragma unroll
    for (int i = 0; i < 8; ++i) dst[tid + 256 * i] = s0[tid + 256 * i];
#pragma unroll
    for (int i = 0; i < 8; ++i) dst[2048 + tid + 256 * i] = s1[tid + 256 * i];
  }
  __syncthreads();

  const int w    = tid >> 5;
  const int lane = tid & 31;
  const int lm   = lane & 15;
  const int lh   = lane >> 4;
  const int rbase = blockIdx.x * 128 + w * 16; // wave's first row (16-aligned)
  const int j    = rbase >> 10;                // r / B (wave-uniform)
  const int rb0  = rbase & (B_DIM - 1);        // r % B (wave-uniform)

  float bias_r[16];
#pragma unroll
  for (int ct = 0; ct < 16; ++ct) bias_r[ct] = bias[ct * 16 + lm];

  v8f cst[4]; // cell state, C-layout: hidden col-tiles 0..3
#pragma unroll
  for (int q = 0; q < 4; ++q) {
#pragma unroll
    for (int e = 0; e < 8; ++e) cst[q][e] = 0.0f;
  }

  v16bf hA[2]; // h as A operand (two K=32 chunks), zero initial state
#pragma unroll
  for (int kc = 0; kc < 2; ++kc)
#pragma unroll
    for (int e = 0; e < 16; ++e) hA[kc][e] = (__bf16)0.0f;

  const int Tp = T_DIM / dil; // T divisible by all dilations
  for (int tp = 0; tp < Tp; ++tp) {
    const int time = tp * dil + j;
    const float* xrow = xin + ((size_t)time * B_DIM + (size_t)(rb0 + lm)) * H_DIM;

    if (time + dil < T_DIM)
      __builtin_prefetch(xrow + (size_t)dil * B_DIM * H_DIM, 0, 3);

    // x tile -> A operand (bf16)
    v16bf xA[2];
#pragma unroll
    for (int kc = 0; kc < 2; ++kc) {
      const int k0 = kc * 32 + lh * 8;
      float4 f0 = *(const float4*)(xrow + k0);
      float4 f1 = *(const float4*)(xrow + k0 + 4);
      float4 f2 = *(const float4*)(xrow + k0 + 16);
      float4 f3 = *(const float4*)(xrow + k0 + 20);
      xA[kc] = packA(f0, f1, f2, f3);
    }

    // gate quadruples: col-tiles {q, q+4, q+8, q+12} = i,f,g,o for hidden
    // units n = q*16 + lane%16 — same lane/VGPR slot across all four.
    // Four independent accumulator chains (XDL ILP); each chain starts from
    // SRC2 = inline 0 so no bias-splat registers/copies are needed.
#pragma unroll
    for (int q = 0; q < 4; ++q) {
      v8f acc[4];
#pragma unroll
      for (int g = 0; g < 4; ++g) {
        const v8f zero = {};
        const v16bf bx0 = *(const v16bf*)&sW[0][q + 4 * g][0][lane][0];
        acc[g] = wmma_bf16(xA[0], bx0, zero);
        const v16bf bh0 = *(const v16bf*)&sW[1][q + 4 * g][0][lane][0];
        acc[g] = wmma_bf16(hA[0], bh0, acc[g]);
        const v16bf bx1 = *(const v16bf*)&sW[0][q + 4 * g][1][lane][0];
        acc[g] = wmma_bf16(xA[1], bx1, acc[g]);
        const v16bf bh1 = *(const v16bf*)&sW[1][q + 4 * g][1][lane][0];
        acc[g] = wmma_bf16(hA[1], bh1, acc[g]);
      }
#pragma unroll
      for (int e = 0; e < 8; ++e) { // C-layout: M = e + 8*lh, N = q*16 + lm
        const float iv = sigf(acc[0][e] + bias_r[q]);
        const float fv = sigf(acc[1][e] + bias_r[q + 4]);
        const float gv = tanhf(acc[2][e] + bias_r[q + 8]);
        const float ov = sigf(acc[3][e] + bias_r[q + 12]);
        const float cv = fv * cst[q][e] + iv * gv;
        cst[q][e] = cv;
        hbuf[w][e + 8 * lh][q * 16 + lm] = ov * tanhf(cv);
      }
    }

    // LDS store -> load turnaround within this wave (DS ops are in-order).
    asm volatile("s_wait_dscnt 0" ::: "memory");

    { // coalesced store of h (+ fused residual on the last cell of group 1)
      float* orow = hout + ((size_t)time * B_DIM + rb0) * H_DIM;
      const float* rrow =
          resid ? resid + ((size_t)time * B_DIM + rb0) * H_DIM : nullptr;
#pragma unroll
      for (int m = 0; m < 16; ++m) {
        float2 hv = *(const float2*)&hbuf[w][m][lane * 2];
        if (rrow) {
          float2 rv = *(const float2*)(rrow + (size_t)m * H_DIM + lane * 2);
          hv.x += rv.x;
          hv.y += rv.y;
        }
        *(float2*)(orow + (size_t)m * H_DIM + lane * 2) = hv;
      }
    }

    // rebuild h A-operand for the next step (pre-residual h, as in reference)
#pragma unroll
    for (int kc = 0; kc < 2; ++kc) {
      const float* hr = &hbuf[w][lm][0];
      const int k0 = kc * 32 + lh * 8;
      float4 f0 = *(const float4*)(hr + k0);
      float4 f1 = *(const float4*)(hr + k0 + 4);
      float4 f2 = *(const float4*)(hr + k0 + 16);
      float4 f3 = *(const float4*)(hr + k0 + 20);
      hA[kc] = packA(f0, f1, f2, f3);
    }
  }
}

// ---------------------------------------------------------------------------
// Head: out = tanh(x @ Wmlp^T + bmlp) @ Wad^T + bad, rows = T*B.
// ---------------------------------------------------------------------------
__global__ __launch_bounds__(256, 4)
void head_kernel(const float* __restrict__ xin,    // (T*B, 64)
                 const __bf16* __restrict__ WmlpP, // [4][2][32][16]
                 const __bf16* __restrict__ WadP,  // [1][2][32][16]
                 const float* __restrict__ bmlp,   // 64
                 const float* __restrict__ bad,    // 12
                 float* __restrict__ out) {        // (T*B, 12)
  __shared__ alignas(32) __bf16 sWm[4][2][32][16];
  __shared__ alignas(32) __bf16 sWa[2][32][16];
  __shared__ alignas(16) float  hbuf[8][16][64];

  const int tid = threadIdx.x;
  {
    uint4* dm = (uint4*)&sWm[0][0][0][0];
    const uint4* sm = (const uint4*)WmlpP;
#pragma unroll
    for (int i = 0; i < 2; ++i) dm[tid + 256 * i] = sm[tid + 256 * i];
    if (tid < 128) ((uint4*)&sWa[0][0][0])[tid] = ((const uint4*)WadP)[tid];
  }
  __syncthreads();

  const int w    = tid >> 5;
  const int lane = tid & 31;
  const int lm   = lane & 15;
  const int lh   = lane >> 4;
  const size_t rbase = (size_t)blockIdx.x * 128 + (size_t)w * 16;

  const float* xrow = xin + (rbase + lm) * H_DIM;
  v16bf xA[2];
#pragma unroll
  for (int kc = 0; kc < 2; ++kc) {
    const int k0 = kc * 32 + lh * 8;
    float4 f0 = *(const float4*)(xrow + k0);
    float4 f1 = *(const float4*)(xrow + k0 + 4);
    float4 f2 = *(const float4*)(xrow + k0 + 16);
    float4 f3 = *(const float4*)(xrow + k0 + 20);
    xA[kc] = packA(f0, f1, f2, f3);
  }

  const float bm = bmlp[lm];          // per-lane: reused with ct offset below
  (void)bm;

#pragma unroll
  for (int ct = 0; ct < 4; ++ct) {
    const v8f zero = {};
    v8f acc;
    {
      const v16bf b0 = *(const v16bf*)&sWm[ct][0][lane][0];
      acc = wmma_bf16(xA[0], b0, zero);
      const v16bf b1 = *(const v16bf*)&sWm[ct][1][lane][0];
      acc = wmma_bf16(xA[1], b1, acc);
    }
    const float bv = bmlp[ct * 16 + lm];
#pragma unroll
    for (int e = 0; e < 8; ++e)
      hbuf[w][e + 8 * lh][ct * 16 + lm] = tanhf(acc[e] + bv);
  }

  asm volatile("s_wait_dscnt 0" ::: "memory");

  v16bf tA[2];
#pragma unroll
  for (int kc = 0; kc < 2; ++kc) {
    const float* hr = &hbuf[w][lm][0];
    const int k0 = kc * 32 + lh * 8;
    float4 f0 = *(const float4*)(hr + k0);
    float4 f1 = *(const float4*)(hr + k0 + 4);
    float4 f2 = *(const float4*)(hr + k0 + 16);
    float4 f3 = *(const float4*)(hr + k0 + 20);
    tA[kc] = packA(f0, f1, f2, f3);
  }

  const v8f zero2 = {};
  v8f acc2;
  {
    const v16bf b0 = *(const v16bf*)&sWa[0][lane][0];
    acc2 = wmma_bf16(tA[0], b0, zero2);
    const v16bf b1 = *(const v16bf*)&sWa[1][lane][0];
    acc2 = wmma_bf16(tA[1], b1, acc2);
  }

  if (lm < O_DIM) {
    const float bv = bad[lm];
#pragma unroll
    for (int e = 0; e < 8; ++e)
      out[(rbase + (size_t)e + 8 * lh) * O_DIM + lm] = acc2[e] + bv;
  }
}

// ---------------------------------------------------------------------------
extern "C" void kernel_launch(void* const* d_in, const int* in_sizes, int n_in,
                              void* d_out, int out_size, void* d_ws, size_t ws_size,
                              hipStream_t stream) {
  (void)in_sizes; (void)n_in; (void)out_size; (void)ws_size;

  const float* y    = (const float*)d_in[0]; // (T,B,64)
  const float* Wih  = (const float*)d_in[1]; // (4,256,64)
  const float* Whh  = (const float*)d_in[2]; // (4,256,64)
  const float* bih  = (const float*)d_in[3]; // (4,256)
  const float* bhh  = (const float*)d_in[4]; // (4,256)
  const float* Wmlp = (const float*)d_in[5]; // (64,64)
  const float* bmlp = (const float*)d_in[6]; // (64)
  const float* Wad  = (const float*)d_in[7]; // (12,64)
  const float* bad  = (const float*)d_in[8]; // (12)
  float* out = (float*)d_out;

  char* ws = (char*)d_ws;
  size_t off = 0;
  auto take = [&](size_t bytes) -> char* {
    char* p = ws + off;
    off = (off + bytes + 255) & ~(size_t)255;
    return p;
  };

  __bf16* WihP  = (__bf16*)take((size_t)4 * 16384 * sizeof(__bf16));
  __bf16* WhhP  = (__bf16*)take((size_t)4 * 16384 * sizeof(__bf16));
  __bf16* WmlpP = (__bf16*)take((size_t)4096 * sizeof(__bf16));
  __bf16* WadP  = (__bf16*)take((size_t)1024 * sizeof(__bf16));
  float*  bsum  = (float*)take((size_t)4 * G_DIM * sizeof(float));
  float*  buf0  = (float*)take((size_t)T_DIM * B_DIM * H_DIM * sizeof(float));
  float*  buf1  = (float*)take((size_t)T_DIM * B_DIM * H_DIM * sizeof(float));

  // Pack weights to WMMA B-operand bf16 layouts; sum the biases.
  for (int c = 0; c < 4; ++c) {
    pack_b_kernel<<<64, 256, 0, stream>>>(Wih + (size_t)c * G_DIM * H_DIM,
                                          WihP + (size_t)c * 16384, G_DIM, 16);
    pack_b_kernel<<<64, 256, 0, stream>>>(Whh + (size_t)c * G_DIM * H_DIM,
                                          WhhP + (size_t)c * 16384, G_DIM, 16);
  }
  pack_b_kernel<<<16, 256, 0, stream>>>(Wmlp, WmlpP, H_DIM, 4);
  pack_b_kernel<<<4, 256, 0, stream>>>(Wad, WadP, O_DIM, 1);
  bias_sum_kernel<<<4, 256, 0, stream>>>(bih, bhh, bsum);

  // Group 0: cells 0 (d=1) and 1 (d=2).
  lstm_scan_kernel<<<(1 * B_DIM) / 128, 256, 0, stream>>>(
      y, nullptr, buf0, WihP + 0 * 16384, WhhP + 0 * 16384, bsum + 0 * G_DIM, 1);
  lstm_scan_kernel<<<(2 * B_DIM) / 128, 256, 0, stream>>>(
      buf0, nullptr, buf1, WihP + 1 * 16384, WhhP + 1 * 16384, bsum + 1 * G_DIM, 2);
  // Group 1: cells 2 (d=4) and 3 (d=8); residual = group-0 output (buf1),
  // fused into cell 3's store (each (t,b) is read-then-written by one wave).
  lstm_scan_kernel<<<(4 * B_DIM) / 128, 256, 0, stream>>>(
      buf1, nullptr, buf0, WihP + 2 * 16384, WhhP + 2 * 16384, bsum + 2 * G_DIM, 4);
  lstm_scan_kernel<<<(8 * B_DIM) / 128, 256, 0, stream>>>(
      buf0, buf1, buf1, WihP + 3 * 16384, WhhP + 3 * 16384, bsum + 3 * G_DIM, 8);

  // Head: tanh MLP + adapter.
  head_kernel<<<(T_DIM * B_DIM) / 128, 256, 0, stream>>>(buf1, WmlpP, WadP,
                                                         bmlp, bad, out);
}
